// Pointnet2SSG_Feature_927712936449
// MI455X (gfx1250) — matrix operand order repaired and checked by reading, not compiled, not run on hardware
//
#include <hip/hip_runtime.h>

typedef __attribute__((ext_vector_type(16))) _Float16 v16h;
typedef __attribute__((ext_vector_type(8)))  _Float16 v8h;
typedef __attribute__((ext_vector_type(8)))  float    v8f;

#define BN_EPS 1e-5f

// ---------------- split pointcloud into xyz0 / feats0 ----------------
__global__ void split_pc(const float* __restrict__ pc, float* __restrict__ xyz0,
                         float* __restrict__ f0, int total /*B*N*/) {
  int i = blockIdx.x * blockDim.x + threadIdx.x;
  if (i >= total) return;
  const float* p = pc + (size_t)i * 6;
  xyz0[i * 3 + 0] = p[0]; xyz0[i * 3 + 1] = p[1]; xyz0[i * 3 + 2] = p[2];
  f0[i * 3 + 0]   = p[3]; f0[i * 3 + 1]   = p[4]; f0[i * 3 + 2]   = p[5];
}

// ---------------- farthest point sampling: one block per batch ----------------
__global__ void fps_kernel(const float* __restrict__ xyz, // [B,N,3]
                           float* __restrict__ mind,      // [B,N] scratch
                           int* __restrict__ out_idx,     // [B,npoint]
                           int N, int npoint) {
  int b = blockIdx.x;
  const float* x = xyz + (size_t)b * N * 3;
  float* md = mind + (size_t)b * N;
  int* oi = out_idx + (size_t)b * npoint;
  __shared__ float sval[256];
  __shared__ int   sidx[256];
  __shared__ float lp[3];
  __shared__ int   slast;
  int t = threadIdx.x;
  for (int p = t; p < N; p += 256) md[p] = 1e10f;
  if (t == 0) { oi[0] = 0; slast = 0; }
  __syncthreads();
  for (int it = 1; it < npoint; ++it) {
    if (t == 0) {
      int l = slast;
      lp[0] = x[l * 3 + 0]; lp[1] = x[l * 3 + 1]; lp[2] = x[l * 3 + 2];
    }
    __syncthreads();
    float lx = lp[0], ly = lp[1], lz = lp[2];
    float best = -1.0f; int bi = 0x7fffffff;
    for (int p = t; p < N; p += 256) {
      float dx = x[p * 3 + 0] - lx, dy = x[p * 3 + 1] - ly, dz = x[p * 3 + 2] - lz;
      float d = dx * dx + dy * dy + dz * dz;
      float m = md[p];
      m = fminf(m, d);
      md[p] = m;
      if (m > best) { best = m; bi = p; }   // ascending p => first max kept
    }
    sval[t] = best; sidx[t] = bi;
    __syncthreads();
    for (int s = 128; s > 0; s >>= 1) {
      if (t < s) {
        float v2 = sval[t + s]; int i2 = sidx[t + s];
        if (v2 > sval[t] || (v2 == sval[t] && i2 < sidx[t])) { sval[t] = v2; sidx[t] = i2; }
      }
      __syncthreads();
    }
    if (t == 0) { slast = sidx[0]; oi[it] = sidx[0]; }
    __syncthreads();
  }
}

// ---------------- gather sampled centers ----------------
__global__ void gather_xyz(const float* __restrict__ xyz, const int* __restrict__ idx,
                           float* __restrict__ nxyz, int total /*B*npoint*/,
                           int N, int npoint) {
  int i = blockIdx.x * blockDim.x + threadIdx.x;
  if (i >= total) return;
  int b = i / npoint;
  int j = idx[i];
  const float* s = xyz + ((size_t)b * N + j) * 3;
  nxyz[i * 3 + 0] = s[0]; nxyz[i * 3 + 1] = s[1]; nxyz[i * 3 + 2] = s[2];
}

// ---------------- ball query: one wave32 per center, ordered-index select ----------------
__global__ void ball_query(const float* __restrict__ xyz, const float* __restrict__ nxyz,
                           int* __restrict__ out, int B, int N, int npoint,
                           float r2, int nsample) {
  int wave = (int)((blockIdx.x * blockDim.x + threadIdx.x) >> 5);
  int lane = threadIdx.x & 31;
  int total = B * npoint;
  if (wave >= total) return;
  int b = wave / npoint;
  const float* cx = nxyz + (size_t)wave * 3;
  float c0 = cx[0], c1 = cx[1], c2 = cx[2];
  const float* x = xyz + (size_t)b * N * 3;
  int* o = out + (size_t)wave * nsample;
  int cnt = 0, first = 0;
  for (int base = 0; base < N && cnt < nsample; base += 32) {   // N is multiple of 32
    int p = base + lane;
    float dx = x[p * 3 + 0] - c0, dy = x[p * 3 + 1] - c1, dz = x[p * 3 + 2] - c2;
    float d = dx * dx + dy * dy + dz * dz;
    bool hit = d < r2;
    unsigned mask = (unsigned)__ballot(hit);   // wave32: low 32 bits
    if (mask) {
      if (cnt == 0) first = base + __builtin_ctz(mask);
      if (hit) {
        int pos = cnt + __builtin_popcount(mask & ((1u << lane) - 1u));
        if (pos < nsample) o[pos] = p;
      }
      cnt += __builtin_popcount(mask);
    }
  }
  if (cnt > nsample) cnt = nsample;
  int fi = (cnt > 0) ? first : 0;
  for (int s = lane; s < nsample; s += 32)
    if (s >= cnt) o[s] = fi;
}

// ------- grouping: centered xyz ++ gathered feats, written as f16, K zero-padded -------
__global__ void group_kernel(const float* __restrict__ xyz, const float* __restrict__ nxyz,
                             const float* __restrict__ feats, const int* __restrict__ idx,
                             _Float16* __restrict__ A16, int total /*B*np*ns*/,
                             int N, int npoint, int nsample, int Cf, int Kpad) {
  int i = blockIdx.x * blockDim.x + threadIdx.x;
  if (i >= total) return;
  int bs = i / nsample;          // b*npoint + s
  int b = bs / npoint;
  int j = idx[i];
  int Cin = 3 + Cf;
  _Float16* row = A16 + (size_t)i * Kpad;
  const float* p = xyz + ((size_t)b * N + j) * 3;
  const float* c = nxyz + (size_t)bs * 3;
  row[0] = (_Float16)(p[0] - c[0]);
  row[1] = (_Float16)(p[1] - c[1]);
  row[2] = (_Float16)(p[2] - c[2]);
  const float* f = feats + ((size_t)b * N + j) * Cf;
  for (int ch = 0; ch < Cf; ++ch) row[3 + ch] = (_Float16)f[ch];
  for (int ch = Cin; ch < Kpad; ++ch) row[ch] = (_Float16)0.0f;
}

// ------- pre-swizzle weights into the WMMA B-fragment layout, f16, K zero-padded -------
// Wf[(((nt*nk + ks)*32 + lane)*16 + e] = f16(W[k*N + n]),
//   k = ks*32 + (lane>>4)*16 + e, n = nt*16 + (lane&15); zero if k >= K.
__global__ void prep_wfrag(const float* __restrict__ W, _Float16* __restrict__ Wf,
                           int K, int N, int Kpad) {
  int i = blockIdx.x * blockDim.x + threadIdx.x;
  int nk = Kpad >> 5;
  int total = (N >> 4) * nk * 512;
  if (i >= total) return;
  int e = i & 15;
  int lane = (i >> 4) & 31;
  int rest = i >> 9;            // nt*nk + ks
  int ks = rest % nk;
  int nt = rest / nk;
  int k = ks * 32 + (lane >> 4) * 16 + e;
  int n = nt * 16 + (lane & 15);
  Wf[i] = (k < K) ? (_Float16)W[(size_t)k * N + n] : (_Float16)0.0f;
}

// ---------------- WMMA GEMM: Y[M,N](f32) = A16[M,Kpad](f16) @ W(f16 fragments) ----------
// One wave computes a 16x32 output tile (2 WMMA accumulators, A reused across 2 n-tiles).
// All loads are unconditional 16-byte vectors; zero guards in the hot loop.
__global__ void wmma_mm(const _Float16* __restrict__ A, const _Float16* __restrict__ Wf,
                        float* __restrict__ Y, int M, int Kpad, int N) {
  int lane = threadIdx.x;                              // 0..31
  int mt = blockIdx.x * blockDim.y + threadIdx.y;      // m-tile (16 rows)
  int ng = blockIdx.y;                                 // n-group (32 cols = 2 tiles)
  int half = lane >> 4;
  int l15 = lane & 15;
  int row = mt * 16 + l15;
  int nk = Kpad >> 5;
  const _Float16* ap  = A + (size_t)row * Kpad + half * 16;
  const _Float16* bp0 = Wf + (((size_t)(2 * ng)     * nk) * 32 + lane) * 16;
  const _Float16* bp1 = Wf + (((size_t)(2 * ng + 1) * nk) * 32 + lane) * 16;
  v8f acc0 = {}, acc1 = {};
  for (int ks = 0; ks < nk; ++ks) {
    // A 16x32 fragment: elements 0..15 -> K = half*16+0..15
    v8h alo = *(const v8h*)(ap);
    v8h ahi = *(const v8h*)(ap + 8);
    v16h a = __builtin_shufflevector(alo, ahi, 0, 1, 2, 3, 4, 5, 6, 7,
                                               8, 9, 10, 11, 12, 13, 14, 15);
    __builtin_prefetch(ap + 32, 0, 1);
    // B fragments: 32 contiguous bytes per lane, fully coalesced
    v8h b0lo = *(const v8h*)(bp0);
    v8h b0hi = *(const v8h*)(bp0 + 8);
    v16h b0 = __builtin_shufflevector(b0lo, b0hi, 0, 1, 2, 3, 4, 5, 6, 7,
                                                  8, 9, 10, 11, 12, 13, 14, 15);
    v8h b1lo = *(const v8h*)(bp1);
    v8h b1hi = *(const v8h*)(bp1 + 8);
    v16h b1 = __builtin_shufflevector(b1lo, b1hi, 0, 1, 2, 3, 4, 5, 6, 7,
                                                  8, 9, 10, 11, 12, 13, 14, 15);
    acc0 = __builtin_amdgcn_wmma_f32_16x16x32_f16(false, a, false, b0,
                                                  (short)0, acc0, false, false);
    acc1 = __builtin_amdgcn_wmma_f32_16x16x32_f16(false, a, false, b1,
                                                  (short)0, acc1, false, false);
    ap  += 32;
    bp0 += 32 * 16;
    bp1 += 32 * 16;
  }
  // C/D layout: VGPR r -> M = r + 8*half ; N = lane&15
  int mb = mt * 16 + half * 8;
  int nc0 = ng * 32 + l15;
  int nc1 = nc0 + 16;
#pragma unroll
  for (int r = 0; r < 8; ++r) {
    Y[(size_t)(mb + r) * N + nc0] = acc0[r];
    Y[(size_t)(mb + r) * N + nc1] = acc1[r];
  }
}

// ---------------- per-channel sum / sumsq (one block per channel) ----------------
__global__ void chan_stats(const float* __restrict__ Y, float* __restrict__ stats,
                           int M, int N) {
  int c = blockIdx.x;
  __shared__ float ss[256], sq[256];
  float s = 0.f, q = 0.f;
  for (int m = threadIdx.x; m < M; m += 256) {
    float v = Y[(size_t)m * N + c];
    s += v; q += v * v;
  }
  ss[threadIdx.x] = s; sq[threadIdx.x] = q;
  __syncthreads();
  for (int st = 128; st > 0; st >>= 1) {
    if (threadIdx.x < st) {
      ss[threadIdx.x] += ss[threadIdx.x + st];
      sq[threadIdx.x] += sq[threadIdx.x + st];
    }
    __syncthreads();
  }
  if (threadIdx.x == 0) { stats[c] = ss[0]; stats[512 + c] = sq[0]; }
}

// ------- batchnorm (population stats) + ReLU: fp32 in place + f16 copy for next GEMM ----
__global__ void bn_relu(float* __restrict__ Y, _Float16* __restrict__ A16,
                        const float* __restrict__ stats,
                        const float* __restrict__ gamma, const float* __restrict__ beta,
                        int M, int N) {
  size_t i = (size_t)blockIdx.x * blockDim.x + threadIdx.x;
  size_t total = (size_t)M * N;
  if (i >= total) return;
  int c = (int)(i % (size_t)N);
  float invM = 1.0f / (float)M;
  float mu = stats[c] * invM;
  float var = stats[512 + c] * invM - mu * mu;
  float v = gamma[c] * (Y[i] - mu) * rsqrtf(var + BN_EPS) + beta[c];
  v = v > 0.f ? v : 0.f;
  Y[i] = v;
  A16[i] = (_Float16)v;   // N is a multiple of 32 => row stride == Kpad of next GEMM
}

// ---------------- max over nsample ----------------
__global__ void maxpool(const float* __restrict__ Y, float* __restrict__ out,
                        int total /*B*np*C*/, int nsample, int C) {
  int i = blockIdx.x * blockDim.x + threadIdx.x;
  if (i >= total) return;
  int c = i % C;
  int bs = i / C;
  const float* base = Y + (size_t)bs * nsample * C + c;
  float m = base[0];
  for (int k = 1; k < nsample; ++k) m = fmaxf(m, base[(size_t)k * C]);
  out[i] = m;
}

extern "C" void kernel_launch(void* const* d_in, const int* in_sizes, int n_in,
                              void* d_out, int out_size, void* d_ws, size_t ws_size,
                              hipStream_t stream) {
  (void)in_sizes; (void)n_in; (void)out_size; (void)ws_size;
  const int B = 8;
  const int NPTS[5] = {16384, 1024, 256, 64, 16};
  const float RAD[4] = {0.02f, 0.04f, 0.06f, 0.08f};
  const int NS[4] = {32, 32, 16, 16};
  const int MLP[4][4] = {{6, 32, 32, 64}, {67, 64, 64, 128},
                         {131, 128, 128, 256}, {259, 256, 256, 512}};
  const int CF[4] = {3, 64, 128, 256};      // feature channels entering layer l
  const int FC[5] = {3, 64, 128, 256, 512}; // l_feats channel counts

  float* out = (float*)d_out;
  size_t off_xyz[5], off_f[5];
  off_xyz[0] = 0;
  for (int l = 1; l < 5; ++l) off_xyz[l] = off_xyz[l - 1] + (size_t)B * NPTS[l - 1] * 3;
  off_f[0] = off_xyz[4] + (size_t)B * NPTS[4] * 3;
  for (int l = 1; l < 5; ++l) off_f[l] = off_f[l - 1] + (size_t)B * NPTS[l - 1] * FC[l - 1];

  // workspace carve-out
  char* wsbase = (char*)d_ws;
  size_t o = 0;
  auto carve = [&](size_t bytes) -> void* {
    void* p = wsbase + o;
    o = (o + bytes + 255) & ~(size_t)255;
    return p;
  };
  float*     mind    = (float*)carve((size_t)B * 16384 * sizeof(float));
  int*       fpsIdx  = (int*)  carve((size_t)B * 1024 * sizeof(int));
  int*       ballIdx = (int*)  carve((size_t)B * 1024 * 32 * sizeof(int));
  _Float16*  A16     = (_Float16*)carve((size_t)262144 * 64 * sizeof(_Float16)); // max M*Kpad
  float*     Ybuf    = (float*)carve((size_t)262144 * 64 * sizeof(float));       // max M*Nc
  _Float16*  Wfrag   = (_Float16*)carve((size_t)512 * 288 * sizeof(_Float16));   // max N*Kpad
  float*     stats   = (float*)carve(1024 * sizeof(float));

  auto kpad32 = [](int k) { return (k + 31) & ~31; };

  const float* pc = (const float*)d_in[0];
  {
    int total = B * NPTS[0];
    split_pc<<<(total + 255) / 256, 256, 0, stream>>>(pc, out + off_xyz[0], out + off_f[0], total);
  }

  for (int l = 0; l < 4; ++l) {
    int N = NPTS[l], np = NPTS[l + 1], ns = NS[l], Cf = CF[l];
    const float* xyz   = out + off_xyz[l];
    const float* feats = out + off_f[l];
    float* nxyz        = out + off_xyz[l + 1];

    fps_kernel<<<B, 256, 0, stream>>>(xyz, mind, fpsIdx, N, np);

    int totalC = B * np;
    gather_xyz<<<(totalC + 255) / 256, 256, 0, stream>>>(xyz, fpsIdx, nxyz, totalC, N, np);

    ball_query<<<(totalC * 32 + 255) / 256, 256, 0, stream>>>(
        xyz, nxyz, ballIdx, B, N, np, RAD[l] * RAD[l], ns);

    int tg = B * np * ns;
    int Kp0 = kpad32(3 + Cf);
    group_kernel<<<(tg + 255) / 256, 256, 0, stream>>>(
        xyz, nxyz, feats, ballIdx, A16, tg, N, np, ns, Cf, Kp0);

    int M = B * np * ns;            // multiple of 64 for all layers
    for (int j = 0; j < 3; ++j) {
      int K = MLP[l][j], Nc = MLP[l][j + 1];
      int Kp = kpad32(K);
      const float* W  = (const float*)d_in[1 + (l * 3 + j) * 3 + 0];
      const float* g  = (const float*)d_in[1 + (l * 3 + j) * 3 + 1];
      const float* bt = (const float*)d_in[1 + (l * 3 + j) * 3 + 2];

      int wtot = (Nc >> 4) * (Kp >> 5) * 512;
      prep_wfrag<<<(wtot + 255) / 256, 256, 0, stream>>>(W, Wfrag, K, Nc, Kp);

      dim3 grid(M / 64, Nc / 32);
      dim3 blk(32, 4);
      wmma_mm<<<grid, blk, 0, stream>>>(A16, Wfrag, Ybuf, M, Kp, Nc);

      chan_stats<<<Nc, 256, 0, stream>>>(Ybuf, stats, M, Nc);

      size_t tot = (size_t)M * Nc;
      bn_relu<<<(unsigned)((tot + 255) / 256), 256, 0, stream>>>(
          Ybuf, A16, stats, g, bt, M, Nc);
    }
    int C3 = MLP[l][3];
    int totalP = B * np * C3;
    maxpool<<<(totalP + 255) / 256, 256, 0, stream>>>(Ybuf, out + off_f[l + 1], totalP, ns, C3);
  }
}
